// SelfAttention_48009144435509
// MI455X (gfx1250) — matrix operand order repaired and checked by reading, not compiled, not run on hardware
//
#include <hip/hip_runtime.h>

typedef __attribute__((ext_vector_type(16))) __bf16 v16bf;
typedef __attribute__((ext_vector_type(8)))  float  v8f;

// Problem constants (from reference setup_inputs)
constexpr int kB  = 16;
constexpr int kT  = 2048;
constexpr int kD  = 256;
constexpr int kBQ = 128;            // query rows per block (8 waves x 16)
constexpr int kSB = 32;             // keys per s-block
constexpr int kNSB = kT / kSB;      // 64
constexpr int kQS = 528;            // padded row stride in bytes (264 bf16) - bank-conflict pad
constexpr int kKRM = kSB * kQS;     // 16896 B row-major K tile
constexpr int kKB  = 16 * 1024;     // 16384 B transposed (B-fragment) K tile

union FragB { v16bf v; uint4 q[2]; };

static __device__ __forceinline__ unsigned pack2(float lo, float hi) {
  union { __bf16 b[2]; unsigned u; } x;
  x.b[0] = (__bf16)lo; x.b[1] = (__bf16)hi;
  return x.u;
}

static __device__ __forceinline__ v8f wmma_bf16(v16bf a, v16bf b, v8f c) {
  return __builtin_amdgcn_wmma_f32_16x16x32_bf16(false, a, false, b, (short)0, c,
                                                 false, false);
}

// Stage one 32x256 K tile: row-major bf16 (for A-fragments of K.Q^T) and
// B-fragment layout (keys packed pairwise per dim column, for P.V).
static __device__ __forceinline__ void stage_k_store(char* Krm, char* KBp, int t,
                                                     const float4* v0, const float4* v1) {
  const int c0 = (t & 63) * 4;
#pragma unroll
  for (int i = 0; i < 4; ++i) {
    const int kp = (t >> 6) + i * 4;
    const int r0 = 2 * kp, r1 = r0 + 1;
    const float4 a = v0[i], b = v1[i];
    *(uint2*)(Krm + r0 * kQS + c0 * 2) = make_uint2(pack2(a.x, a.y), pack2(a.z, a.w));
    *(uint2*)(Krm + r1 * kQS + c0 * 2) = make_uint2(pack2(b.x, b.y), pack2(b.z, b.w));
    const float av[4] = {a.x, a.y, a.z, a.w};
    const float bv[4] = {b.x, b.y, b.z, b.w};
#pragma unroll
    for (int j = 0; j < 4; ++j) {
      const int c = c0 + j;
      const int laneT = (c & 15) + 16 * ((r0 >> 3) & 1);
      *(unsigned*)(KBp + (c >> 4) * 1024 + laneT * 32 + ((r0 & 7) + 8 * (r0 >> 4)) * 2) =
          pack2(av[j], bv[j]);
    }
  }
}

static __device__ __forceinline__ void stage_k_load(const float* Qb, int s0, int t,
                                                    float4* v0, float4* v1) {
  const int c0 = (t & 63) * 4;
#pragma unroll
  for (int i = 0; i < 4; ++i) {
    const int kp = (t >> 6) + i * 4;
    v0[i] = *(const float4*)(Qb + (size_t)(s0 + 2 * kp) * kD + c0);
    v1[i] = *(const float4*)(Qb + (size_t)(s0 + 2 * kp + 1) * kD + c0);
  }
}

__global__ __launch_bounds__(256)
void SelfAttention_fa_bf16_wmma(const float* __restrict__ Q,
                                const int*   __restrict__ qlen,
                                const float* __restrict__ biasp,
                                float*       __restrict__ Out) {
  __shared__ __attribute__((aligned(16))) char sQ[kBQ * kQS];     // 67584
  __shared__ __attribute__((aligned(16))) char sKrm[2][kKRM];     // 2x16896
  __shared__ __attribute__((aligned(16))) char sKB[2][kKB];       // 2x16384
  __shared__ float sInv[8][16];

  const int t    = threadIdx.x;
  const int lane = t & 31;
  const int wave = t >> 5;
  const int hi   = lane >> 4;   // 0/1
  const int ln   = lane & 15;

  const int b     = blockIdx.x >> 4;        // 16 m-blocks per batch
  const int tBase = (blockIdx.x & 15) * kBQ;

  const float* Qb = Q + (size_t)b * kT * kD;
  const int len   = qlen[b];
  const float bias = biasp[0];

  // keys with s >= len contribute exp(-1e29) == 0: skip whole key-blocks.
  const int nsb = (len + kSB - 1) / kSB;    // 1..kNSB (len >= 1)

  // ---- stage the 128x256 query tile as row-major bf16 (padded stride) ----
#pragma unroll 4
  for (int i = 0; i < 32; ++i) {
    const int id  = t + 256 * i;        // 0..8191 float4 ids
    const int row = id >> 6;
    const int c4  = id & 63;
    const float4 v = *(const float4*)(Qb + (size_t)(tBase + row) * kD + c4 * 4);
    *(uint2*)(sQ + row * kQS + c4 * 8) = make_uint2(pack2(v.x, v.y), pack2(v.z, v.w));
  }

  // ---- stage first K s-block ----
  {
    float4 v0[4], v1[4];
    stage_k_load(Qb, 0, t, v0, v1);
    stage_k_store(sKrm[0], sKB[0], t, v0, v1);
  }
  __syncthreads();

  // ---- hoist loop-invariant Q B-fragments (this wave's 16 query rows) ----
  FragB bq[8];
#pragma unroll
  for (int kk = 0; kk < 8; ++kk) {
    const char* br = sQ + (wave * 16 + ln) * kQS + kk * 64 + hi * 16;
    bq[kk].q[0] = *(const uint4*)(br);
    bq[kk].q[1] = *(const uint4*)(br + 32);
  }

  const v8f vz = {0.f, 0.f, 0.f, 0.f, 0.f, 0.f, 0.f, 0.f};
  v8f acc[16];
#pragma unroll
  for (int n = 0; n < 16; ++n) acc[n] = vz;
  float rs = 0.f;

  const int qg = tBase + wave * 16 + ln;    // this lane's query row (as N col)

#pragma unroll 1
  for (int sb = 0; sb < nsb; ++sb) {
    const int cur = sb & 1;
    // prefetch next tile (global loads issued before compute)
    float4 pv0[4], pv1[4];
    if (sb + 1 < nsb) stage_k_load(Qb, (sb + 1) * kSB, t, pv0, pv1);

    const char* Krm = sKrm[cur];
    const char* KBp = sKB[cur];
    const int s0 = sb * kSB;

    // ---- S^T = K . Q^T  (two 16-key tiles, K-dim = 256 in 8 steps) ----
    v8f S0 = vz, S1 = vz;
#pragma unroll
    for (int kk = 0; kk < 8; ++kk) {
      FragB a0, a1;
      const char* ar0 = Krm + ln * kQS + kk * 64 + hi * 16;
      a0.q[0] = *(const uint4*)(ar0);
      a0.q[1] = *(const uint4*)(ar0 + 32);
      const char* ar1 = ar0 + 16 * kQS;
      a1.q[0] = *(const uint4*)(ar1);
      a1.q[1] = *(const uint4*)(ar1 + 32);
      S0 = wmma_bf16(a0.v, bq[kk].v, S0);
      S1 = wmma_bf16(a1.v, bq[kk].v, S1);
    }

    // ---- unstabilized exp (+ masks only where needed) + rowsum ----
    // tail block (s >= len inside) or diagonal overlap with [tBase, tBase+128)?
    const bool needMask = (s0 + kSB > len) |
                          ((s0 < tBase + kBQ) & (s0 + kSB > tBase));
    float e0[8], e1[8];
    if (needMask) {
#pragma unroll
      for (int r = 0; r < 8; ++r) {
        const int k0 = s0 + r + 8 * hi;     // keys 0..15 tile
        const int k1 = k0 + 16;             // keys 16..31 tile
        const bool m0 = (k0 == qg) || (k0 >= len);
        const bool m1 = (k1 == qg) || (k1 >= len);
        e0[r] = m0 ? 0.f : __expf(S0[r]);
        e1[r] = m1 ? 0.f : __expf(S1[r]);
      }
    } else {
#pragma unroll
      for (int r = 0; r < 8; ++r) {
        e0[r] = __expf(S0[r]);
        e1[r] = __expf(S1[r]);
      }
    }
    v16bf pa;
#pragma unroll
    for (int r = 0; r < 8; ++r) {
      rs += e0[r] + e1[r];
      pa[r]     = (__bf16)e0[r];            // keys 0-7 / 8-15 (lane-half)
      pa[8 + r] = (__bf16)e1[r];            // keys 16-23 / 24-31
    }

    // ---- acc += P . V   (V = same K tile, B-fragment layout) ----
#pragma unroll
    for (int n = 0; n < 16; ++n) {
      FragB bv;
      const char* p = KBp + n * 1024 + lane * 32;
      bv.q[0] = *(const uint4*)(p);
      bv.q[1] = *(const uint4*)(p + 16);
      acc[n] = wmma_bf16(pa, bv.v, acc[n]);
    }

    // ---- write prefetched tile into the other buffer ----
    if (sb + 1 < nsb)
      stage_k_store((char*)sKrm[cur ^ 1], (char*)sKB[cur ^ 1], t, pv0, pv1);
    __syncthreads();
  }

  // ---- finalize: rowsum across lane halves, denom, scaled store ----
  const float rtot = rs + __shfl_xor(rs, 16, 32);
  if (lane < 16) sInv[wave][lane] = 1.0f / (rtot + __expf(bias));
  __syncthreads();

  float iv[8];
#pragma unroll
  for (int r = 0; r < 8; ++r) iv[r] = sInv[wave][8 * hi + r];

  float* Ob = Out + (size_t)b * kT * kD;
#pragma unroll
  for (int n = 0; n < 16; ++n) {
#pragma unroll
    for (int r = 0; r < 8; ++r) {
      const int qrow = tBase + wave * 16 + 8 * hi + r;
      Ob[(size_t)qrow * kD + 16 * n + ln] = acc[n][r] * iv[r];
    }
  }
}

extern "C" void kernel_launch(void* const* d_in, const int* in_sizes, int n_in,
                              void* d_out, int out_size, void* d_ws, size_t ws_size,
                              hipStream_t stream) {
  const float* q    = (const float*)d_in[0];
  const int*   qlen = (const int*)d_in[1];
  const float* bias = (const float*)d_in[2];
  float*       out  = (float*)d_out;

  dim3 grid(kB * (kT / kBQ));   // 16 * 16 = 256 blocks
  dim3 block(256);              // 8 wave32 waves
  hipLaunchKernelGGL(SelfAttention_fa_bf16_wmma, grid, block, 0, stream,
                     q, qlen, bias, out);
}
